// EncodeProcessDecode_9165460209751
// MI455X (gfx1250) — matrix operand order repaired and checked by reading, not compiled, not run on hardware
//
#include <hip/hip_runtime.h>
#include <hip/hip_bf16.h>

// ---------------------------------------------------------------------------
// EncodeProcessDecode GNN for MI455X (gfx1250, wave32).
// All GEMMs use v_wmma_f32_16x16x32_f16 (f16 in, fp32 accumulate).
// Block tile: 32 rows x 128 cols; each wave computes two stacked 16x16 tiles
// that share one B fragment (2 WMMAs per B fetch -> halved L2 weight traffic).
// ---------------------------------------------------------------------------

typedef _Float16 half_t;
typedef __attribute__((ext_vector_type(16))) _Float16 v16h;
typedef __attribute__((ext_vector_type(8)))  _Float16 v8h;
typedef __attribute__((ext_vector_type(8)))  float    v8f;

#define NN 10000      // nodes
#define NE 100000     // edges

static inline int cdiv(int a, int b) { return (a + b - 1) / b; }

// ---------------- utility kernels ----------------

// W is (K x 128) fp32 row-major; write W^T as (128 x K) f16 row-major so that
// the WMMA B-fragment per lane is two 16-byte runs along K (k 8h.., 16+8h..).
__global__ void k_wt_transpose(const float* __restrict__ W, half_t* __restrict__ WT,
                               int K, int N) {
  int i = blockIdx.x * 256 + threadIdx.x;
  if (i < K * N) {
    int k = i / N, n = i % N;
    WT[(size_t)n * K + k] = (half_t)W[i];
  }
}

__global__ void k_f32_to_f16(const float* __restrict__ in, half_t* __restrict__ out, int n) {
  int i = blockIdx.x * 256 + threadIdx.x;
  if (i < n) out[i] = (half_t)in[i];
}

// cat_e[ed] = [ x[dst[ed]], x[src[ed]], e[ed] ]   (f16, width 384)
__global__ void k_cat_edge(const half_t* __restrict__ xf, const half_t* __restrict__ ef,
                           const int* __restrict__ src, const int* __restrict__ dst,
                           half_t* __restrict__ cat) {
  int i = blockIdx.x * 256 + threadIdx.x;
  if (i >= NE * 384) return;
  int ed = i / 384, c = i % 384;
  half_t v;
  if (c < 128)      v = xf[(size_t)dst[ed] * 128 + c];
  else if (c < 256) v = xf[(size_t)src[ed] * 128 + (c - 128)];
  else              v = ef[(size_t)ed * 128 + (c - 256)];
  cat[i] = v;
}

// cat_n[i] = [ x[i], agg[i] ]   (f16, width 256)
__global__ void k_cat_node(const half_t* __restrict__ xf, const float* __restrict__ agg,
                           half_t* __restrict__ cat) {
  int i = blockIdx.x * 256 + threadIdx.x;
  if (i >= NN * 256) return;
  int nd = i / 256, c = i % 256;
  cat[i] = (c < 128) ? xf[(size_t)nd * 128 + c] : (half_t)agg[(size_t)nd * 128 + (c - 128)];
}

// agg = segment_sum(e_new, dst)  via fp32 global atomics
__global__ void k_scatter_add(const float* __restrict__ e_new, const int* __restrict__ dst,
                              float* __restrict__ agg) {
  int i = blockIdx.x * 256 + threadIdx.x;
  if (i >= NE * 128) return;
  int ed = i >> 7, c = i & 127;
  atomicAdd(&agg[(size_t)dst[ed] * 128 + c], e_new[i]);
}

// base += delta (fp32), and refresh the f16 mirror used by the next GEMM
__global__ void k_residual(float* __restrict__ base, const float* __restrict__ delta,
                           half_t* __restrict__ out16, int n) {
  int i = blockIdx.x * 256 + threadIdx.x;
  if (i < n) {
    float v = base[i] + delta[i];
    base[i] = v;
    out16[i] = (half_t)v;
  }
}

// ---------------- fused MLP layer (WMMA GEMM + bias + ReLU | LayerNorm) -----
//
// Block = 256 threads = 8 waves. Block tile: 32 rows x 128 cols (full output
// width, so the final LayerNorm is done in-block). Wave w owns columns
// [16w, 16w+16) and two row subtiles (rows 0-15 and 16-31 of the strip).
//
// A fragment (16x32 f16, ISA 7.12.2): lane L<16 holds row M=L, K {0..7} in
// v0..3 and {16..23} in v4..7; lane L+16 holds K {8..15} and {24..31}.
// -> two contiguous 16B chunks per lane, read from an LDS-staged A strip
//    (row stride K+8 halves: 16B-aligned, lane bank stride 4 -> no conflicts).
// B fragment (32x16 f16, per ISA B-layout): lane holds column N=lane%16,
// K {0..7,16..23} (lanes 0-15) or {8..15,24..31} (lanes 16-31) -> two 16B chunks
// of W^T[n][k0...], loaded from global and reused by both row subtiles.
template <int K, bool LN>
__global__ __launch_bounds__(256, 2) void k_mlp_layer(
    const half_t* __restrict__ A, int M,
    const half_t* __restrict__ WT,       // 128 x K, f16
    const float* __restrict__ bias,      // 128
    const float* __restrict__ gamma,     // 128 (LN only)
    const float* __restrict__ beta,      // 128 (LN only)
    float* __restrict__ out32,           // M x 128 (LN only, may be null)
    half_t* __restrict__ out16)          // M x 128 (may be null in LN mode)
{
  constexpr int KP = K + 8;              // padded LDS stride in halves (16B-aligned rows)
  __shared__ __align__(16) half_t sA[32 * KP];
  __shared__ __align__(16) float  sC[LN ? 32 * 132 : 16];

  const int tid  = threadIdx.x;
  const int lane = tid & 31;
  const int wave = tid >> 5;
  const int row0 = blockIdx.x * 32;

  // Cooperative 16-byte-chunk load of the 32xK A strip into LDS.
  constexpr int CPR = K / 8;             // 16B chunks per row
  for (int c = tid; c < 32 * CPR; c += 256) {
    int m = c / CPR, kc = (c % CPR) * 8;
    uint4 v = {0u, 0u, 0u, 0u};
    int row = row0 + m;
    if (row < M) v = *(const uint4*)(A + (size_t)row * K + kc);
    *(uint4*)(&sA[m * KP + kc]) = v;
  }
  __syncthreads();

  const int col = lane & 15;
  const int hi  = lane >> 4;
  const int n   = wave * 16 + col;       // output column owned by this lane

  v8f acc0 = {};                         // rows row0 + 0..15
  v8f acc1 = {};                         // rows row0 + 16..31
#pragma unroll
  for (int k0 = 0; k0 < K; k0 += 32) {
    // B fragment: two 16B chunks of W^T row n (K 8h..8h+7, 16+8h..), shared by both subtiles
    const half_t* pb = WT + (size_t)n * K + k0 + hi * 8;
    v8h b0 = *(const v8h*)(pb);
    v8h b1 = *(const v8h*)(pb + 16);
    // A fragments for the two row subtiles
    const half_t* pa0 = &sA[col * KP + k0 + hi * 8];
    const half_t* pa1 = &sA[(16 + col) * KP + k0 + hi * 8];
    v8h a00 = *(const v8h*)(pa0);
    v8h a01 = *(const v8h*)(pa0 + 16);
    v8h a10 = *(const v8h*)(pa1);
    v8h a11 = *(const v8h*)(pa1 + 16);
    v16h av0, av1, bv;
#pragma unroll
    for (int i = 0; i < 8; ++i) {
      av0[i] = a00[i]; av0[8 + i] = a01[i];
      av1[i] = a10[i]; av1[8 + i] = a11[i];
      bv[i]  = b0[i];  bv[8 + i]  = b1[i];
    }
    acc0 = __builtin_amdgcn_wmma_f32_16x16x32_f16(
        false, av0, false, bv, (short)0, acc0, false, false);
    acc1 = __builtin_amdgcn_wmma_f32_16x16x32_f16(
        false, av1, false, bv, (short)0, acc1, false, false);
  }

  const float bn = bias[n];
  if (!LN) {
    // hidden layer: bias + ReLU, emit f16 for the next GEMM
#pragma unroll
    for (int r = 0; r < 8; ++r) {
      int m = r + hi * 8;
      int rowA = row0 + m, rowB = row0 + 16 + m;
      if (rowA < M) {
        float v = acc0[r] + bn;
        out16[(size_t)rowA * 128 + n] = (half_t)(v > 0.f ? v : 0.f);
      }
      if (rowB < M) {
        float v = acc1[r] + bn;
        out16[(size_t)rowB * 128 + n] = (half_t)(v > 0.f ? v : 0.f);
      }
    }
  } else {
    // final layer: bias, then 128-wide LayerNorm across the block tile.
    // C/D layout (ISA 7.12.2): c[r] is row (r + 8*hi), column n.
#pragma unroll
    for (int r = 0; r < 8; ++r) {
      sC[(r + hi * 8) * 132 + n]        = acc0[r] + bn;
      sC[(16 + r + hi * 8) * 132 + n]   = acc1[r] + bn;
    }
    __syncthreads();

    // 8 threads per row, 16 columns each; the 8 threads of a row form a
    // contiguous 8-lane group inside one wave32 -> xor-shuffle reduce.
    const int rowl = tid >> 3, seg = tid & 7;
    float s = 0.f, ss = 0.f;
#pragma unroll
    for (int j = 0; j < 16; ++j) {
      float v = sC[rowl * 132 + seg * 16 + j];
      s += v; ss += v * v;
    }
#pragma unroll
    for (int off = 1; off < 8; off <<= 1) {
      s  += __shfl_xor(s,  off, 32);
      ss += __shfl_xor(ss, off, 32);
    }
    const float mu = s * (1.f / 128.f);
    const float rs = rsqrtf(ss * (1.f / 128.f) - mu * mu + 1e-5f);
    const int row = row0 + rowl;
    if (row < M) {
#pragma unroll
      for (int j = 0; j < 16; ++j) {
        int nn = seg * 16 + j;
        float o = (sC[rowl * 132 + nn] - mu) * rs * gamma[nn] + beta[nn];
        if (out32) out32[(size_t)row * 128 + nn] = o;
        if (out16) out16[(size_t)row * 128 + nn] = (half_t)o;
      }
    }
  }
}

// ---------------------------------------------------------------------------

extern "C" void kernel_launch(void* const* d_in, const int* in_sizes, int n_in,
                              void* d_out, int out_size, void* d_ws, size_t ws_size,
                              hipStream_t stream) {
  (void)in_sizes; (void)n_in; (void)out_size; (void)ws_size;

  const float* x_in  = (const float*)d_in[0];
  const int*   eidx  = (const int*)d_in[1];
  const float* ef_in = (const float*)d_in[2];
  const int* src = eidx;        // edge_index[0]
  const int* dst = eidx + NE;   // edge_index[1]

  // Flattened param order: enc_node, enc_edge, gnn[s].{edge,node}, dec_node, dec_edge
  // Each MLP: W0,W1,W2,b0,b1,b2,g,b  (8 arrays)
  const int mlp_base[16] = {3, 11, 19, 27, 35, 43, 51, 59, 67, 75, 83, 91, 99, 107, 115, 123};
  const int mlp_k0[16]   = {128, 128, 384, 256, 384, 256, 384, 256,
                            384, 256, 384, 256, 384, 256, 128, 128};

  // ---- carve workspace ----
  char* ws = (char*)d_ws;
  size_t off = 0;
  auto alloc = [&](size_t bytes) -> void* {
    void* p = ws + off;
    off = (off + bytes + 255) & ~(size_t)255;
    return p;
  };

  half_t* wtp[16][3];
  for (int mi = 0; mi < 16; ++mi)
    for (int l = 0; l < 3; ++l) {
      int Kd = (l == 0) ? mlp_k0[mi] : 128;
      wtp[mi][l] = (half_t*)alloc((size_t)Kd * 128 * 2);
    }

  half_t* xin16 = (half_t*)alloc((size_t)NN * 128 * 2);
  half_t* ein16 = (half_t*)alloc((size_t)NE * 128 * 2);
  half_t* x16   = (half_t*)alloc((size_t)NN * 128 * 2);
  half_t* e16   = (half_t*)alloc((size_t)NE * 128 * 2);
  half_t* h1    = (half_t*)alloc((size_t)NE * 128 * 2);
  half_t* h2    = (half_t*)alloc((size_t)NE * 128 * 2);
  half_t* cat_e = (half_t*)alloc((size_t)NE * 384 * 2);
  half_t* cat_n = (half_t*)alloc((size_t)NN * 256 * 2);
  float*  xbuf  = (float*)alloc((size_t)NN * 128 * 4);
  float*  ebuf  = (float*)alloc((size_t)NE * 128 * 4);
  float*  enew  = (float*)alloc((size_t)NE * 128 * 4);
  float*  xnew  = (float*)alloc((size_t)NN * 128 * 4);
  float*  agg   = (float*)alloc((size_t)NN * 128 * 4);

  // ---- weight prep: fp32 (KxN) -> f16 transposed (NxK) ----
  for (int mi = 0; mi < 16; ++mi)
    for (int l = 0; l < 3; ++l) {
      int Kd = (l == 0) ? mlp_k0[mi] : 128;
      int sz = Kd * 128;
      k_wt_transpose<<<cdiv(sz, 256), 256, 0, stream>>>(
          (const float*)d_in[mlp_base[mi] + l], wtp[mi][l], Kd, 128);
    }

  k_f32_to_f16<<<cdiv(NN * 128, 256), 256, 0, stream>>>(x_in, xin16, NN * 128);
  k_f32_to_f16<<<cdiv(NE * 128, 256), 256, 0, stream>>>(ef_in, ein16, NE * 128);

  auto launch_layer = [&](int K, bool ln, const half_t* Ain, int M, const half_t* WT,
                          const float* b, const float* g, const float* bt,
                          float* o32, half_t* o16) {
    dim3 grid(cdiv(M, 32)), blk(256);
    if (ln) {
      k_mlp_layer<128, true><<<grid, blk, 0, stream>>>(Ain, M, WT, b, g, bt, o32, o16);
    } else {
      switch (K) {
        case 128: k_mlp_layer<128, false><<<grid, blk, 0, stream>>>(Ain, M, WT, b, g, bt, o32, o16); break;
        case 256: k_mlp_layer<256, false><<<grid, blk, 0, stream>>>(Ain, M, WT, b, g, bt, o32, o16); break;
        default:  k_mlp_layer<384, false><<<grid, blk, 0, stream>>>(Ain, M, WT, b, g, bt, o32, o16); break;
      }
    }
  };

  auto run_mlp = [&](int mi, const half_t* Ain, int M, int Kin, float* o32, half_t* o16) {
    int base = mlp_base[mi];
    const float* b0 = (const float*)d_in[base + 3];
    const float* b1 = (const float*)d_in[base + 4];
    const float* b2 = (const float*)d_in[base + 5];
    const float* g  = (const float*)d_in[base + 6];
    const float* bt = (const float*)d_in[base + 7];
    launch_layer(Kin, false, Ain, M, wtp[mi][0], b0, nullptr, nullptr, nullptr, h1);
    launch_layer(128, false, h1,  M, wtp[mi][1], b1, nullptr, nullptr, nullptr, h2);
    launch_layer(128, true,  h2,  M, wtp[mi][2], b2, g, bt, o32, o16);
  };

  // ---- encode ----
  run_mlp(0, xin16, NN, 128, xbuf, x16);   // enc_node -> x
  run_mlp(1, ein16, NE, 128, ebuf, e16);   // enc_edge -> e

  // ---- process: 6 residual message-passing steps ----
  for (int s = 0; s < 6; ++s) {
    k_cat_edge<<<cdiv(NE * 384, 256), 256, 0, stream>>>(x16, e16, src, dst, cat_e);
    run_mlp(2 + 2 * s, cat_e, NE, 384, enew, nullptr);          // e_new (fp32)

    hipMemsetAsync(agg, 0, (size_t)NN * 128 * 4, stream);
    k_scatter_add<<<cdiv(NE * 128, 256), 256, 0, stream>>>(enew, dst, agg);

    k_cat_node<<<cdiv(NN * 256, 256), 256, 0, stream>>>(x16, agg, cat_n);
    run_mlp(3 + 2 * s, cat_n, NN, 256, xnew, nullptr);          // x_new (fp32)

    k_residual<<<cdiv(NN * 128, 256), 256, 0, stream>>>(xbuf, xnew, x16, NN * 128);
    k_residual<<<cdiv(NE * 128, 256), 256, 0, stream>>>(ebuf, enew, e16, NE * 128);
  }

  // ---- decode: outputs concatenated (node part, then edge part), fp32 ----
  float* outn = (float*)d_out;
  float* oute = outn + (size_t)NN * 128;
  run_mlp(14, x16, NN, 128, outn, nullptr);
  run_mlp(15, e16, NE, 128, oute, nullptr);
}
